// TopKSparseAttention_27736898797637
// MI455X (gfx1250) — matrix operand order repaired and checked by reading, not compiled, not run on hardware
//
#include <hip/hip_runtime.h>
#include <hip/hip_bf16.h>

// ---------------- problem constants ----------------
constexpr int Bc = 4;        // batch
constexpr int Cc = 256;      // channels
constexpr int Nn = 4096;     // tokens = H*W
constexpr int KKEEP = 409;   // int(0.1 * 4096)

// ---------------- CDNA5 WMMA types ----------------
typedef __attribute__((ext_vector_type(16))) __bf16 vbf16x16;
typedef __attribute__((ext_vector_type(8)))  __bf16 vbf16x8;
typedef __attribute__((ext_vector_type(8)))  float  vf32x8;

union AFrag {
    vbf16x16 v;
    vbf16x8  h[2];
};

__device__ __forceinline__ unsigned short f2bf(float f) {
    unsigned int u = __float_as_uint(f);
    unsigned int r = u + 0x7FFFu + ((u >> 16) & 1u);   // round-to-nearest-even
    return (unsigned short)(r >> 16);
}

// monotonic float -> uint key (ascending order preserved)
__device__ __forceinline__ unsigned int ordkey(float f) {
    unsigned int u = __float_as_uint(f);
    return u ^ ((unsigned int)(((int)u) >> 31) | 0x80000000u);
}

__device__ __forceinline__ vf32x8 wmma_bf16(vbf16x16 a, vbf16x16 b, vf32x8 c) {
    return __builtin_amdgcn_wmma_f32_16x16x32_bf16(false, a, false, b,
                                                   (short)0, c, false, false);
}

// ---- CDNA5 async global->LDS copy (ASYNCcnt path, no VGPR round-trip) ----
// Per-lane: copy 16 bytes MEM[gaddr] -> LDS[lds_off]. LDS offset is the low
// 32 bits of the generic LDS pointer.
__device__ __forceinline__ void async_copy_b128(unsigned lds_off, const void* gptr) {
    asm volatile("global_load_async_to_lds_b128 %0, %1, off"
                 :
                 : "v"(lds_off), "v"((unsigned long long)(size_t)gptr)
                 : "memory");
}
__device__ __forceinline__ void wait_async() {
    asm volatile("s_wait_asynccnt 0" ::: "memory");
}

// ---------------- kernel 1a: x [B,C,N] fp32 -> Xbf [B,N,C] bf16 ----------------
__global__ void cvt_x_kernel(const float* __restrict__ x,
                             unsigned short* __restrict__ Xbf) {
    size_t i = (size_t)blockIdx.x * blockDim.x + threadIdx.x;
    if (i >= (size_t)Bc * Cc * Nn) return;
    int n = (int)(i % Nn);
    size_t t = i / Nn;
    int c = (int)(t % Cc);
    int b = (int)(t / Cc);
    Xbf[((size_t)b * Nn + n) * Cc + c] = f2bf(x[i]);
}

// ---------------- kernel 1b: W[d][c] fp32 -> Wt[c][d] bf16 (q,k,v) ----------------
__global__ void cvt_w_kernel(const float* __restrict__ Wq,
                             const float* __restrict__ Wk,
                             const float* __restrict__ Wv,
                             unsigned short* __restrict__ Wtq,
                             unsigned short* __restrict__ Wtk,
                             unsigned short* __restrict__ Wtv) {
    int i = blockIdx.x * blockDim.x + threadIdx.x;
    if (i >= 3 * Cc * Cc) return;
    int w = i / (Cc * Cc);
    int r = i % (Cc * Cc);
    int d = r / Cc;
    int c = r % Cc;
    const float* W = (w == 0) ? Wq : (w == 1) ? Wk : Wv;
    unsigned short* Wt = (w == 0) ? Wtq : (w == 1) ? Wtk : Wtv;
    Wt[c * Cc + d] = f2bf(W[d * Cc + c]);
}

// ---------------- kernel 2: QKV projection GEMM (WMMA bf16) ----------------
// block = 256 threads (8 waves), one block per 16-token row tile.
// Q stored scaled by 1/sqrt(C); K stored transposed Kt[b][c][n]; V row-major.
__global__ void qkv_gemm_kernel(const unsigned short* __restrict__ Xbf,
                                const unsigned short* __restrict__ Wtq,
                                const unsigned short* __restrict__ Wtk,
                                const unsigned short* __restrict__ Wtv,
                                unsigned short* __restrict__ Qbf,
                                unsigned short* __restrict__ Ktbf,
                                unsigned short* __restrict__ Vbf) {
    __shared__ __align__(16) unsigned short xs[16 * Cc];     // 8 KB

    const int tid  = threadIdx.x;
    const int lane = tid & 31;
    const int wv   = tid >> 5;
    const int rt   = blockIdx.x;               // row tile 0..1023
    const int b    = rt >> 8;
    const int n0   = (rt & 255) << 4;          // token row within batch
    const size_t g0 = (size_t)rt << 4;         // global token row (b*N + n0)

    // stage 16x256 bf16 X tile via async global->LDS (2 x B128 per thread)
    {
        const unsigned short* src = Xbf + g0 * Cc;
        const unsigned ldsbase = (unsigned)(size_t)(void*)xs;
        async_copy_b128(ldsbase + tid * 16,        src + tid * 8);
        async_copy_b128(ldsbase + 4096 + tid * 16, src + 2048 + tid * 8);
        wait_async();
    }
    __syncthreads();

    // hoist A fragments (shared by all 48 output tiles of this block)
    const int m    = lane & 15;
    const int half = lane >> 4;
    AFrag af[8];
    for (int ks = 0; ks < 8; ++ks) {
        const unsigned short* base = xs + m * Cc + ks * 32 + half * 8;
        af[ks].h[0] = *(const vbf16x8*)base;
        af[ks].h[1] = *(const vbf16x8*)(base + 16);
    }

    // 48 tiles = 3 mats x 16 col tiles; 6 per wave
    for (int t = wv * 6; t < wv * 6 + 6; ++t) {
        const int mat = t >> 4;
        const int jb  = (t & 15) << 4;
        const unsigned short* Wt = (mat == 0) ? Wtq : (mat == 1) ? Wtk : Wtv;

        vf32x8 acc = {0.f, 0.f, 0.f, 0.f, 0.f, 0.f, 0.f, 0.f};
        for (int ks = 0; ks < 8; ++ks) {
            const unsigned short* bp = Wt + (ks * 32 + lane) * Cc + jb;
            __builtin_prefetch(bp + 32 * Cc, 0, 3);
            vbf16x16 bf;
            for (int s = 0; s < 16; ++s) {
                const int kk = ks * 32 + half * 8 + (s & 7) + ((s >> 3) << 4);
                const unsigned short u = Wt[kk * Cc + jb + m];
                bf[s] = __builtin_bit_cast(__bf16, u);
            }
            acc = wmma_bf16(af[ks].v, bf, acc);
        }
        for (int r = 0; r < 8; ++r) {
            const int mm = r + (half << 3);
            const int d  = jb + m;
            const float v = acc[r];
            if (mat == 0) {
                Qbf[((size_t)g0 + mm) * Cc + d] = f2bf(v * 0.0625f); // 1/sqrt(256)
            } else if (mat == 1) {
                Ktbf[((size_t)b * Cc + d) * Nn + (n0 + mm)] = f2bf(v);
            } else {
                Vbf[((size_t)g0 + mm) * Cc + d] = f2bf(v);
            }
        }
    }
}

// ---------------- kernel 3: scores + top-k radix select + softmax + AV ----------------
// block = 512 threads (16 wave32), one block per (batch, 16-query tile).
// Dynamic LDS layout (319744 B total, needs CDNA5's 320KB/WGP):
//   [0      , 262144): sc   float[16][4096]  score tile
//   [262144 , 294912): at   bf16 [16][4096]  attn weights
//   [294912 , 311296): hist u32  [16][256]   radix histograms
//   [311296 , 319488): qs   bf16 [16][256]   Q tile
//   [319488 , 319744): rowmax/rowsum/thr
constexpr unsigned SMEM3 = 319744;

__global__ void attn_kernel(const unsigned short* __restrict__ Qbf,
                            const unsigned short* __restrict__ Ktbf,
                            const unsigned short* __restrict__ Vbf,
                            float* __restrict__ out) {
    extern __shared__ __align__(16) unsigned char smem[];
    float*          sc     = (float*)smem;
    unsigned short* at     = (unsigned short*)(smem + 262144);
    unsigned int*   hist   = (unsigned int*)(smem + 294912);
    unsigned short* qs     = (unsigned short*)(smem + 311296);
    float*          rowmax = (float*)(smem + 319488);
    float*          rowsum = (float*)(smem + 319552);
    unsigned int*   thrb   = (unsigned int*)(smem + 319616);

    const int tid  = threadIdx.x;
    const int lane = tid & 31;
    const int wv   = tid >> 5;                 // wave id 0..15
    const int blk  = blockIdx.x;
    const int b    = blk >> 8;
    const int q0   = (blk & 255) << 4;
    const size_t g0 = (size_t)blk << 4;        // b*N + q0

    // ---- load 16x256 Q tile via async global->LDS (1 x B128 per thread) ----
    {
        const unsigned short* src = Qbf + g0 * Cc;
        const unsigned ldsbase = (unsigned)(size_t)(void*)qs;
        async_copy_b128(ldsbase + tid * 16, src + tid * 8);
        wait_async();
    }
    __syncthreads();

    const int m    = lane & 15;
    const int half = lane >> 4;

    // hoisted A fragments of the Q tile (reused by 16 score tiles per wave)
    AFrag aq[8];
    for (int ks = 0; ks < 8; ++ks) {
        const unsigned short* base = qs + m * Cc + ks * 32 + half * 8;
        aq[ks].h[0] = *(const vbf16x8*)base;
        aq[ks].h[1] = *(const vbf16x8*)(base + 16);
    }

    // ---- phase 1: scores = (Q/sqrtC) @ K^T   (16 x 4096, WMMA) ----
    for (int t = 0; t < 16; ++t) {
        const int jb = ((wv << 4) + t) << 4;   // key column base
        vf32x8 acc = {0.f, 0.f, 0.f, 0.f, 0.f, 0.f, 0.f, 0.f};
        for (int ks = 0; ks < 8; ++ks) {
            const int kc = (ks << 5) + lane;
            const unsigned short* bp = Ktbf + ((size_t)b * Cc + kc) * Nn + jb;
            vbf16x16 bf;
            for (int s = 0; s < 16; ++s) {
                const int kk = (ks << 5) + half * 8 + (s & 7) + ((s >> 3) << 4);
                const unsigned short u = Ktbf[((size_t)b * Cc + kk) * Nn + jb + m];
                bf[s] = __builtin_bit_cast(__bf16, u);
            }
            acc = wmma_bf16(aq[ks].v, bf, acc);
        }
        for (int r = 0; r < 8; ++r) {
            const int mm = r + (half << 3);
            sc[mm * Nn + jb + m] = acc[r];
        }
    }
    __syncthreads();

    // ---- phase 2: per-row max (wave wv owns row wv) ----
    {
        float mx = -3.4e38f;
        for (int i = lane; i < Nn; i += 32) mx = fmaxf(mx, sc[wv * Nn + i]);
        for (int off = 16; off > 0; off >>= 1) mx = fmaxf(mx, __shfl_xor(mx, off, 32));
        if (lane == 0) { rowmax[wv] = mx; thrb[wv] = 0u; }
    }
    __syncthreads();

    // ---- phase 3: radix select 409th-largest per row (4 byte passes) ----
    int kneed = KKEEP;
    for (int p = 3; p >= 0; --p) {
        for (int bin = lane; bin < 256; bin += 32) hist[(wv << 8) + bin] = 0u;
        __syncthreads();
        const unsigned int pref = thrb[wv];
        for (int i = lane; i < Nn; i += 32) {
            const unsigned int u = ordkey(sc[wv * Nn + i]);
            const bool match = (p == 3) ||
                ((u >> ((p + 1) * 8)) == (pref >> ((p + 1) * 8)));
            if (match)
                atomicAdd(&hist[(wv << 8) + ((u >> (p * 8)) & 255u)], 1u);
        }
        __syncthreads();
        if (lane == 0) {
            int cum = 0, sel = 0;
            for (int bin = 255; bin >= 0; --bin) {
                const int c = (int)hist[(wv << 8) + bin];
                if (cum + c >= kneed) { sel = bin; break; }
                cum += c;
            }
            kneed -= cum;
            thrb[wv] = pref | ((unsigned int)sel << (p * 8));
        }
        __syncthreads();
    }

    // ---- phase 4: masked softmax (keep keys with key >= threshold) ----
    {
        const float rm = rowmax[wv];
        const unsigned int th = thrb[wv];
        float s = 0.f;
        for (int i = lane; i < Nn; i += 32) {
            const float f = sc[wv * Nn + i];
            const float e = (ordkey(f) >= th) ? __expf(f - rm) : 0.f;
            sc[wv * Nn + i] = e;
            s += e;
        }
        for (int off = 16; off > 0; off >>= 1) s += __shfl_xor(s, off, 32);
        if (lane == 0) rowsum[wv] = s;
    }
    __syncthreads();
    for (int idx = tid; idx < 16 * Nn; idx += 512) {
        const int r = idx >> 12;
        at[idx] = f2bf(sc[idx] / rowsum[r]);
    }
    __syncthreads();

    // ---- phase 5: out = attn @ V   (16 x 256, WMMA over K=4096) ----
    {
        const int jb = wv << 4;                // each wave one 16-wide d tile
        vf32x8 acc = {0.f, 0.f, 0.f, 0.f, 0.f, 0.f, 0.f, 0.f};
        for (int ks = 0; ks < 128; ++ks) {
            const int kb = ks << 5;
            const unsigned short* ap = at + m * Nn + kb + half * 8;
            AFrag afr;
            afr.h[0] = *(const vbf16x8*)ap;
            afr.h[1] = *(const vbf16x8*)(ap + 16);
            const unsigned short* bp = Vbf + ((size_t)b * Nn + kb + lane) * Cc + jb;
            __builtin_prefetch(bp + 32 * Cc, 0, 3);
            vbf16x16 bf;
            for (int s = 0; s < 16; ++s) {
                const int kk = kb + half * 8 + (s & 7) + ((s >> 3) << 4);
                const unsigned short u = Vbf[((size_t)b * Nn + kk) * Cc + jb + m];
                bf[s] = __builtin_bit_cast(__bf16, u);
            }
            acc = wmma_bf16(afr.v, bf, acc);
        }
        for (int r = 0; r < 8; ++r) {
            const int mm = r + (half << 3);
            const int d  = jb + m;
            out[((size_t)b * Cc + d) * Nn + (q0 + mm)] = acc[r];
        }
    }
}

// ---------------- host-side launch ----------------
extern "C" void kernel_launch(void* const* d_in, const int* in_sizes, int n_in,
                              void* d_out, int out_size, void* d_ws, size_t ws_size,
                              hipStream_t stream) {
    const float* x  = (const float*)d_in[0];
    const float* Wq = (const float*)d_in[1];
    const float* Wk = (const float*)d_in[2];
    const float* Wv = (const float*)d_in[3];
    float* out = (float*)d_out;

    unsigned char* ws = (unsigned char*)d_ws;
    const size_t SZ_X  = (size_t)Bc * Nn * Cc * 2;   // 8 MB bf16
    const size_t SZ_W  = (size_t)Cc * Cc * 2;        // 128 KB bf16
    unsigned short* Xbf  = (unsigned short*)(ws);
    unsigned short* Wtq  = (unsigned short*)(ws + SZ_X);
    unsigned short* Wtk  = (unsigned short*)(ws + SZ_X + SZ_W);
    unsigned short* Wtv  = (unsigned short*)(ws + SZ_X + 2 * SZ_W);
    unsigned short* Qbf  = (unsigned short*)(ws + SZ_X + 3 * SZ_W);
    unsigned short* Ktbf = (unsigned short*)(ws + 2 * SZ_X + 3 * SZ_W);
    unsigned short* Vbf  = (unsigned short*)(ws + 3 * SZ_X + 3 * SZ_W);

    // 1) convert / transpose inputs to bf16
    {
        const size_t total = (size_t)Bc * Cc * Nn;
        cvt_x_kernel<<<(unsigned)((total + 255) / 256), 256, 0, stream>>>(x, Xbf);
        cvt_w_kernel<<<(3 * Cc * Cc + 255) / 256, 256, 0, stream>>>(
            Wq, Wk, Wv, Wtq, Wtk, Wtv);
    }
    // 2) QKV projection (WMMA); 1024 row tiles
    qkv_gemm_kernel<<<(Bc * Nn) / 16, 256, 0, stream>>>(
        Xbf, Wtq, Wtk, Wtv, Qbf, Ktbf, Vbf);
    // 3) scores + top-k + softmax + AV (WMMA); 1024 query tiles
    attn_kernel<<<(Bc * Nn) / 16, 512, SMEM3, stream>>>(Qbf, Ktbf, Vbf, out);
}
